// SparseMultiheadAttention_17575006175530
// MI455X (gfx1250) — matrix operand order repaired and checked by reading, not compiled, not run on hardware
//
#include <hip/hip_runtime.h>
#include <hip/hip_bf16.h>
#include <stdint.h>

// ---------------- problem constants ----------------
#define S_DIM 2048
#define B_DIM 2
#define C_DIM 1024
#define H_DIM 16
#define CC    64
#define EPQ   32
#define E_DIM (S_DIM * EPQ)          // 65536 edges
#define M_DIM (S_DIM * B_DIM)        // 4096 GEMM rows

// GEMM macro-tiling: block = 256 threads (8 waves), 128x64 block tile,
// each wave computes one 16x64 strip (4 accumulators), K-step = 32.
#define MBLK 128
#define NBLK 64
#define KSTEP 32
#define NK (C_DIM / KSTEP)

// ---------------- vector types ----------------
typedef __attribute__((ext_vector_type(16))) __bf16       v16bf;
typedef __attribute__((ext_vector_type(8)))  float        v8f;
typedef __attribute__((ext_vector_type(4)))  unsigned int v4u;
typedef __attribute__((ext_vector_type(4)))  int          v4i;

// ---------------- async-copy availability probe ----------------
#if defined(__AMDGCN__) && __has_builtin(__builtin_amdgcn_global_load_async_to_lds_b128)
#define ASYNC_OK 1
#else
#define ASYNC_OK 0
#endif

// 16-byte global -> LDS copy (async if available, else load+ds_store).
// Builtin signature (from hipcc diagnostic): param0 = AS1 (global) v4i*,
// so: (global src, lds dst, imm offset, imm cpol).
__device__ __forceinline__ void cp16_g2l(const uint16_t* g, uint16_t* l) {
#if ASYNC_OK
    __builtin_amdgcn_global_load_async_to_lds_b128(
        (__attribute__((address_space(1))) v4i*)(uintptr_t)(g),
        (__attribute__((address_space(3))) v4i*)(uint32_t)(uintptr_t)(l),
        0, 0);
#else
    *(v4u*)l = *(const v4u*)g;
#endif
}

template <int N>
__device__ __forceinline__ void wait_async() {
#if ASYNC_OK
#if __has_builtin(__builtin_amdgcn_s_wait_asynccnt)
    __builtin_amdgcn_s_wait_asynccnt(N);
#else
    asm volatile("s_wait_asynccnt %0" ::"i"(N) : "memory");
#endif
#endif
}

// ---------------- bf16 helpers (round-to-nearest-even split) ----------------
__device__ __forceinline__ uint16_t f2bf(float f) {
    unsigned u = __float_as_uint(f);
    unsigned r = u + 0x7FFFu + ((u >> 16) & 1u);
    return (uint16_t)(r >> 16);
}
__device__ __forceinline__ float bf2f(uint16_t h) {
    return __uint_as_float(((unsigned)h) << 16);
}

__device__ __forceinline__ void atomicMaxFloat(float* addr, float val) {
    if (val >= 0.f) atomicMax((int*)addr, __float_as_int(val));
    else            atomicMin((unsigned int*)addr, __float_as_uint(val));
}

// ---------------- kernel: fp32 -> bf16 hi/lo split ----------------
__global__ void k_cvt_bf16x2(const float* __restrict__ src,
                             uint16_t* __restrict__ hi,
                             uint16_t* __restrict__ lo, int n) {
    int i = blockIdx.x * blockDim.x + threadIdx.x;
    if (i < n) {
        float v = src[i];
        uint16_t h = f2bf(v);
        hi[i] = h;
        lo[i] = f2bf(v - bf2f(h));
    }
}

// ---------------- kernel: init global max scalar ----------------
__global__ void k_init_max(float* gmax) {
    *gmax = __uint_as_float(0xFF800000u); // -inf
}

// ---------------- kernel: LDS-staged bf16x3 WMMA GEMM ----------------
// out[m,n] = sum_k A[m,k] * W[n,k] + bias[n]   (torch Linear: x @ W.T + b)
// A: M_DIM x C_DIM bf16 hi/lo; W: (N,K) row-major per matrix; z selects matrix.
__global__ void __launch_bounds__(256)
k_gemm_bf16x3(const uint16_t* __restrict__ Ah,
              const uint16_t* __restrict__ Al,
              const uint16_t* __restrict__ Wh,
              const uint16_t* __restrict__ Wl,
              const float* __restrict__ bias0,
              const float* __restrict__ bias1,
              const float* __restrict__ bias2,
              float* __restrict__ outBase,
              int outStride) {
    // double-buffered staging panels (48 KB of the WGP's 320 KB LDS)
    __shared__ alignas(16) uint16_t sAh[2 * MBLK * KSTEP];
    __shared__ alignas(16) uint16_t sAl[2 * MBLK * KSTEP];
    __shared__ alignas(16) uint16_t sBh[2 * NBLK * KSTEP];
    __shared__ alignas(16) uint16_t sBl[2 * NBLK * KSTEP];

    const int z = blockIdx.z;
    const uint16_t* wh = Wh + (size_t)z * (C_DIM * C_DIM);
    const uint16_t* wl = Wl + (size_t)z * (C_DIM * C_DIM);
    const float* bias  = (z == 0) ? bias0 : ((z == 1) ? bias1 : bias2);
    float* out = outBase + (size_t)z * outStride;

    const int tid  = threadIdx.x;
    const int lane = tid & 31;
    const int wave = tid >> 5;
    const int n0 = blockIdx.x * NBLK;
    const int m0 = blockIdx.y * MBLK;

    // staging map: A panel 128x32 (hi & lo) = 512 16B chunks -> 2/thread each;
    //              B panel 64x32 (hi & lo)  = 256 16B chunks -> 1/thread each.
    const int cA0 = tid * 2, cA1 = tid * 2 + 1;
    const int rA0 = cA0 >> 2, oA0 = (cA0 & 3) * 8;
    const int rA1 = cA1 >> 2, oA1 = (cA1 & 3) * 8;
    const int rB  = tid >> 2, oB  = (tid & 3) * 8;

    const uint16_t* gAh = Ah + (size_t)(m0 + rA0) * C_DIM + oA0;
    const uint16_t* gAh2= Ah + (size_t)(m0 + rA1) * C_DIM + oA1;
    const uint16_t* gAl = Al + (size_t)(m0 + rA0) * C_DIM + oA0;
    const uint16_t* gAl2= Al + (size_t)(m0 + rA1) * C_DIM + oA1;
    const uint16_t* gBh = wh + (size_t)(n0 + rB) * C_DIM + oB;
    const uint16_t* gBl = wl + (size_t)(n0 + rB) * C_DIM + oB;

    auto stage = [&](int k0, int d) {
        const int dA = d * (MBLK * KSTEP);
        const int dB = d * (NBLK * KSTEP);
        cp16_g2l(gAh  + k0, sAh + dA + rA0 * KSTEP + oA0);
        cp16_g2l(gAh2 + k0, sAh + dA + rA1 * KSTEP + oA1);
        cp16_g2l(gAl  + k0, sAl + dA + rA0 * KSTEP + oA0);
        cp16_g2l(gAl2 + k0, sAl + dA + rA1 * KSTEP + oA1);
        cp16_g2l(gBh  + k0, sBh + dB + rB * KSTEP + oB);
        cp16_g2l(gBl  + k0, sBl + dB + rB * KSTEP + oB);
    };

    // ISA 16-bit A (16x32) frag: lanes 0-15 row=lane, K={0..7,16..23};
    // lanes 16-31 row=lane-16, K={8..15,24..31}.
    const int aRow = 16 * wave + (lane & 15);
    const int kA   = (lane < 16) ? 0 : 8;
    // ISA 16-bit B (32x16) frag: lanes 0-15 col=lane,K={0..7,16..23}; lanes 16-31 K={8..15,24..31}.
    const int bCol = lane & 15;
    const int kB   = (lane < 16) ? 0 : 8;

    v8f acc[4];
#pragma unroll
    for (int t = 0; t < 4; ++t) acc[t] = v8f{0.f,0.f,0.f,0.f,0.f,0.f,0.f,0.f};

    stage(0, 0);

    for (int kb = 0; kb < NK; ++kb) {
        const int d = kb & 1;
        if (kb + 1 < NK) {
            stage((kb + 1) * KSTEP, d ^ 1);
            wait_async<6>();   // most recent 6 = next stage; current stage done
        } else {
            wait_async<0>();
        }
        __syncthreads();       // current panels visible to all waves

        const uint16_t* pAh = sAh + d * (MBLK * KSTEP) + aRow * KSTEP + kA;
        const uint16_t* pAl = sAl + d * (MBLK * KSTEP) + aRow * KSTEP + kA;
        v16bf fAh, fAl;
        {
            v4u a0 = *(const v4u*)(pAh);
            v4u a1 = *(const v4u*)(pAh + 16);
            v4u l0 = *(const v4u*)(pAl);
            v4u l1 = *(const v4u*)(pAl + 16);
            fAh = __builtin_bit_cast(v16bf, __builtin_shufflevector(a0, a1, 0,1,2,3,4,5,6,7));
            fAl = __builtin_bit_cast(v16bf, __builtin_shufflevector(l0, l1, 0,1,2,3,4,5,6,7));
        }

#pragma unroll
        for (int t = 0; t < 4; ++t) {
            const int bRow = t * 16 + bCol;
            const uint16_t* pBh = sBh + d * (NBLK * KSTEP) + bRow * KSTEP + kB;
            const uint16_t* pBl = sBl + d * (NBLK * KSTEP) + bRow * KSTEP + kB;
            v4u b0 = *(const v4u*)(pBh);
            v4u b1 = *(const v4u*)(pBh + 16);
            v4u c0 = *(const v4u*)(pBl);
            v4u c1 = *(const v4u*)(pBl + 16);
            v16bf fBh = __builtin_bit_cast(v16bf, __builtin_shufflevector(b0, b1, 0,1,2,3,4,5,6,7));
            v16bf fBl = __builtin_bit_cast(v16bf, __builtin_shufflevector(c0, c1, 0,1,2,3,4,5,6,7));

            // bf16x3 compensation: hi*hi + hi*lo + lo*hi, fp32 accumulate
            acc[t] = __builtin_amdgcn_wmma_f32_16x16x32_bf16(false, fAl, false, fBh, (short)0, acc[t], false, false);
            acc[t] = __builtin_amdgcn_wmma_f32_16x16x32_bf16(false, fAh, false, fBl, (short)0, acc[t], false, false);
            acc[t] = __builtin_amdgcn_wmma_f32_16x16x32_bf16(false, fAh, false, fBh, (short)0, acc[t], false, false);
        }
        __syncthreads();       // all waves done reading before next overwrite
    }

    // C/D layout: VGPR r holds row 16*wave + r + (lane<16 ? 0 : 8), col = lane&15
    const int rbase = m0 + 16 * wave + ((lane >> 4) * 8);
#pragma unroll
    for (int t = 0; t < 4; ++t) {
        const int col = n0 + t * 16 + (lane & 15);
        const float bv = bias[col];
#pragma unroll
        for (int r = 0; r < 8; ++r)
            out[(size_t)(rbase + r) * C_DIM + col] = acc[t][r] + bv;
    }
}

// ---------------- kernel: per-edge scaled dot products + global max ----------------
// One wave per (edge, b). qk stored as [B*H][E] for coalesced reads later.
__global__ void k_edge_qk(const float* __restrict__ q,
                          const float* __restrict__ k,
                          const int* __restrict__ q_id,
                          const int* __restrict__ k_id,
                          float* __restrict__ qk,
                          float* __restrict__ gmax) {
    __shared__ float smax[8];
    const int lane = threadIdx.x & 31;
    const int wave = threadIdx.x >> 5;
    const int task = blockIdx.x * 8 + wave;  // E*B tasks, exact multiple
    const int e = task >> 1;
    const int b = task & 1;

    const float* qrow = q + ((size_t)q_id[e] * B_DIM + b) * C_DIM;
    const float* krow = k + ((size_t)k_id[e] * B_DIM + b) * C_DIM;

    float wmax = __uint_as_float(0xFF800000u);
#pragma unroll
    for (int h = 0; h < H_DIM; ++h) {
        const int base = h * CC;
        float p = qrow[base + lane]      * krow[base + lane]
                + qrow[base + 32 + lane] * krow[base + 32 + lane];
#pragma unroll
        for (int off = 16; off; off >>= 1) p += __shfl_xor(p, off, 32);
        p *= 0.125f;  // 1/sqrt(64)
        if (lane == 0) qk[(size_t)(b * H_DIM + h) * E_DIM + e] = p;
        wmax = fmaxf(wmax, p);
    }
    if (lane == 0) smax[wave] = wmax;
    __syncthreads();
    if (threadIdx.x == 0) {
        float m = smax[0];
#pragma unroll
        for (int i = 1; i < 8; ++i) m = fmaxf(m, smax[i]);
        atomicMaxFloat(gmax, m);
    }
}

// ---------------- kernel: softmax + weighted-value gather ----------------
// Edges for query s are exactly [32s, 32s+32) (q_id = repeat(arange(S), 32)),
// so segment sums are wave-local: no atomics. One wave per (s, b, h).
// Writes attention output directly as bf16 hi/lo for the final WMMA GEMM.
__global__ void k_softmax_av(const float* __restrict__ qk,
                             const float* __restrict__ v,
                             const int* __restrict__ k_id,
                             const float* __restrict__ gmax,
                             uint16_t* __restrict__ attn_h,
                             uint16_t* __restrict__ attn_l) {
    const int lane = threadIdx.x & 31;
    const int wave = threadIdx.x >> 5;
    const int task = blockIdx.x * 8 + wave;      // S*B*H tasks, exact multiple
    const int h = task & (H_DIM - 1);
    const int b = (task >> 4) & (B_DIM - 1);
    const int s = task >> 5;

    const float mx = *gmax;
    const int e = s * EPQ + lane;
    const int kid = k_id[e];
    const float ew = expf(qk[(size_t)(b * H_DIM + h) * E_DIM + e] - mx);

    float d = ew;
#pragma unroll
    for (int off = 16; off; off >>= 1) d += __shfl_xor(d, off, 32);

    float acc0 = 0.f, acc1 = 0.f;
#pragma unroll 4
    for (int j = 0; j < EPQ; ++j) {
        const float w  = __shfl(ew, j, 32);
        const int   kj = __shfl(kid, j, 32);
        const float* vrow = v + ((size_t)kj * B_DIM + b) * C_DIM + h * CC;
        acc0 += w * vrow[lane];
        acc1 += w * vrow[lane + 32];
    }
    const float inv_d = 1.f / d;
    const float r0 = acc0 * inv_d;
    const float r1 = acc1 * inv_d;

    const size_t oidx = ((size_t)s * B_DIM + b) * C_DIM + h * CC + lane;
    uint16_t h0 = f2bf(r0);
    attn_h[oidx] = h0;
    attn_l[oidx] = f2bf(r0 - bf2f(h0));
    uint16_t h1 = f2bf(r1);
    attn_h[oidx + 32] = h1;
    attn_l[oidx + 32] = f2bf(r1 - bf2f(h1));
}

// ---------------- launcher ----------------
extern "C" void kernel_launch(void* const* d_in, const int* in_sizes, int n_in,
                              void* d_out, int out_size, void* d_ws, size_t ws_size,
                              hipStream_t stream) {
    (void)in_sizes; (void)n_in; (void)out_size; (void)ws_size;

    const float* x    = (const float*)d_in[0];
    const int*   q_id = (const int*)  d_in[1];
    const int*   k_id = (const int*)  d_in[2];
    const float* Wq   = (const float*)d_in[3];
    const float* bq   = (const float*)d_in[4];
    const float* Wk   = (const float*)d_in[5];
    const float* bk   = (const float*)d_in[6];
    const float* Wv   = (const float*)d_in[7];
    const float* bv   = (const float*)d_in[8];
    const float* Wx   = (const float*)d_in[9];
    const float* bx   = (const float*)d_in[10];
    float* out = (float*)d_out;

    // workspace carve-up (256B aligned)
    char* ws = (char*)d_ws;
    size_t off = 0;
    auto carve = [&](size_t bytes) -> char* {
        char* p = ws + off;
        off += (bytes + 255) & ~(size_t)255;
        return p;
    };
    const size_t nX = (size_t)M_DIM * C_DIM;   // 4M elems
    const size_t nW = (size_t)C_DIM * C_DIM;   // 1M elems per matrix

    uint16_t* xh    = (uint16_t*)carve(nX * 2);
    uint16_t* xl    = (uint16_t*)carve(nX * 2);
    uint16_t* Wh    = (uint16_t*)carve(4 * nW * 2);   // [Wq|Wk|Wv|Wx]
    uint16_t* Wl    = (uint16_t*)carve(4 * nW * 2);
    float*    qkv   = (float*)   carve(3 * nX * 4);   // [q|k|v] fp32
    float*    qkbuf = (float*)   carve((size_t)B_DIM * H_DIM * E_DIM * 4);
    uint16_t* ath   = (uint16_t*)carve(nX * 2);
    uint16_t* atl   = (uint16_t*)carve(nX * 2);
    float*    gmax  = (float*)   carve(256);

    // 1) fp32 -> bf16 hi/lo splits
    k_cvt_bf16x2<<<(int)(nX / 256), 256, 0, stream>>>(x,  xh, xl, (int)nX);
    k_cvt_bf16x2<<<(int)(nW / 256), 256, 0, stream>>>(Wq, Wh + 0 * nW, Wl + 0 * nW, (int)nW);
    k_cvt_bf16x2<<<(int)(nW / 256), 256, 0, stream>>>(Wk, Wh + 1 * nW, Wl + 1 * nW, (int)nW);
    k_cvt_bf16x2<<<(int)(nW / 256), 256, 0, stream>>>(Wv, Wh + 2 * nW, Wl + 2 * nW, (int)nW);
    k_cvt_bf16x2<<<(int)(nW / 256), 256, 0, stream>>>(Wx, Wh + 3 * nW, Wl + 3 * nW, (int)nW);

    // 2) fused QKV projection (z = 0,1,2 -> q,k,v)
    {
        dim3 grid(C_DIM / NBLK, M_DIM / MBLK, 3);
        k_gemm_bf16x3<<<grid, 256, 0, stream>>>(
            xh, xl, Wh, Wl, bq, bk, bv, qkv, (int)nX);
    }
    const float* qmat = qkv + 0 * nX;
    const float* kmat = qkv + 1 * nX;
    const float* vmat = qkv + 2 * nX;

    // 3) per-edge dots + global max
    k_init_max<<<1, 1, 0, stream>>>(gmax);
    k_edge_qk<<<(E_DIM * B_DIM) / 8, 256, 0, stream>>>(qmat, kmat, q_id, k_id, qkbuf, gmax);

    // 4) softmax + AV gather -> attention output (bf16 hi/lo)
    k_softmax_av<<<(S_DIM * B_DIM * H_DIM) / 8, 256, 0, stream>>>(
        qkbuf, vmat, k_id, gmax, ath, atl);

    // 5) final projection -> d_out (fp32)
    {
        dim3 grid(C_DIM / NBLK, M_DIM / MBLK, 1);
        k_gemm_bf16x3<<<grid, 256, 0, stream>>>(
            ath, atl, Wh + 3 * nW, Wl + 3 * nW, bx, bx, bx, out, 0);
    }
}
